// NGCF_18657337934507
// MI455X (gfx1250) — compile-verified, matrix operand order checked by reading, not run on hardware
//
#include <hip/hip_runtime.h>
#include <hip/hip_bf16.h>

typedef __attribute__((ext_vector_type(16))) __bf16 v16bf;
typedef __attribute__((ext_vector_type(8)))  float  v8f;

// ---------------- helpers ----------------
__device__ inline unsigned short f2bf(float f) {
    unsigned int b = __float_as_uint(f);
    unsigned int r = b + 0x7FFFu + ((b >> 16) & 1u);   // round-to-nearest-even
    return (unsigned short)(r >> 16);
}
__device__ inline float leaky02(float x) { return x > 0.f ? x : 0.2f * x; }

__device__ inline void atomicMaxFloat(float* addr, float val) {
    unsigned int* ai = (unsigned int*)addr;
    unsigned int old = *ai;
    while (__uint_as_float(old) < val) {
        unsigned int assumed = old;
        old = atomicCAS(ai, assumed, __float_as_uint(val));
        if (old == assumed) break;
    }
}

union FragBF { v16bf v; uint4 q[2]; };

// ---------------- WMMA GEMM: C[M,N] = act(A[M,K] * Bt[N,K]^T + bias + addend) ----------------
// A fp32 (converted to bf16 into LDS), Bt bf16 N-major (pre-transposed weights; loaded
// into LDS via CDNA5 async global->LDS DMA, tracked by ASYNCcnt).
// 256 threads = 8 wave32. Block tile 128x128, wave tile 32x64, K-step 32.
#define LSTR 40   // LDS row stride in bf16 elements (80B: 16B-aligned, conflict-broken)

__global__ __launch_bounds__(256)
void gemm_bf16_wmma(const float* __restrict__ A, const unsigned short* __restrict__ Bt,
                    const float* __restrict__ bias, const float* __restrict__ addend,
                    float* __restrict__ C, int M, int N, int K, int act) {
    __shared__ __align__(16) unsigned short sA[128 * LSTR];
    __shared__ __align__(16) unsigned short sB[128 * LSTR];

    const int tid  = threadIdx.x;
    const int wave = tid >> 5;
    const int lane = tid & 31;
    const int lm   = lane & 15;
    const int lh   = lane >> 4;          // which K-half-of-16 this lane holds
    const int wm   = wave >> 1;          // 0..3 : 32-row strip
    const int wn   = wave & 1;           // 0..1 : 64-col strip
    const int m0   = blockIdx.y * 128;
    const int n0   = blockIdx.x * 128;

    v8f zero = {0.f,0.f,0.f,0.f,0.f,0.f,0.f,0.f};
    v8f acc[2][4];
    #pragma unroll
    for (int i = 0; i < 2; ++i)
        #pragma unroll
        for (int s = 0; s < 4; ++s) acc[i][s] = zero;

    for (int k0 = 0; k0 < K; k0 += 32) {
        // cooperative load: 128 rows x 32 cols per tile, 8 elements per (thread, iter)
        #pragma unroll
        for (int it = 0; it < 2; ++it) {
            int idx = it * 256 + tid;          // 0..511
            int r   = idx >> 2;                // 0..127
            int cg  = (idx & 3) * 8;           // 0,8,16,24
            // A tile: fp32 -> bf16 convert through VGPRs
            {
                const float* ap = A + (size_t)(m0 + r) * K + k0 + cg;
                float4 f0 = *(const float4*)(ap);
                float4 f1 = *(const float4*)(ap + 4);
                union { unsigned short u[8]; uint4 q; } pk;
                pk.u[0] = f2bf(f0.x); pk.u[1] = f2bf(f0.y);
                pk.u[2] = f2bf(f0.z); pk.u[3] = f2bf(f0.w);
                pk.u[4] = f2bf(f1.x); pk.u[5] = f2bf(f1.y);
                pk.u[6] = f2bf(f1.z); pk.u[7] = f2bf(f1.w);
                *(uint4*)&sA[r * LSTR + cg] = pk.q;
            }
            // B tile: bf16 rows of Bt (N-major) -> async DMA straight into LDS
            {
                int n = n0 + r;
                unsigned short* lp = &sB[r * LSTR + cg];
                if (n < N) {
                    const unsigned short* gp = Bt + (size_t)n * K + k0 + cg;
                    unsigned lds_off = (unsigned)(uintptr_t)lp;  // low 32b of flat = DS addr
                    asm volatile("global_load_async_to_lds_b128 %0, %1, off"
                                 :: "v"(lds_off), "v"(gp) : "memory");
                } else {
                    *(uint4*)lp = make_uint4(0u, 0u, 0u, 0u);
                }
            }
        }
        asm volatile("s_wait_asynccnt 0x0" ::: "memory");
        __syncthreads();

        FragBF a[2];
        #pragma unroll
        for (int i = 0; i < 2; ++i) {
            const unsigned short* p = &sA[(wm * 32 + i * 16 + lm) * LSTR + lh * 8];
            a[i].q[0] = *(const uint4*)(p);
            a[i].q[1] = *(const uint4*)(p + 16);
        }
        #pragma unroll
        for (int s = 0; s < 4; ++s) {
            FragBF b;
            const unsigned short* p = &sB[(wn * 64 + s * 16 + lm) * LSTR + lh * 8];
            b.q[0] = *(const uint4*)(p);
            b.q[1] = *(const uint4*)(p + 16);
            #pragma unroll
            for (int i = 0; i < 2; ++i)
                acc[i][s] = __builtin_amdgcn_wmma_f32_16x16x32_bf16(
                    false, a[i].v, false, b.v, (short)0, acc[i][s], false, false);
        }
        __syncthreads();
    }

    // epilogue: C frag layout -> VGPR r holds row (r + 8*lh), col = lm within 16x16 tile
    #pragma unroll
    for (int s = 0; s < 4; ++s) {
        int col = n0 + wn * 64 + s * 16 + lm;
        if (col >= N) continue;
        float bv = bias ? bias[col] : 0.f;
        #pragma unroll
        for (int i = 0; i < 2; ++i) {
            #pragma unroll
            for (int r = 0; r < 8; ++r) {
                int row = m0 + wm * 32 + i * 16 + lh * 8 + r;
                float v = acc[i][s][r] + bv;
                if (addend) v += addend[(size_t)row * N + col];
                if (act == 1) v = fmaxf(v, 0.f);
                C[(size_t)row * N + col] = v;
            }
        }
    }
}

// ---------------- misc kernels ----------------
__global__ void transpose_cvt_kernel(const float* __restrict__ W, unsigned short* __restrict__ Wt,
                                     int K, int N) {   // W[K][N] -> Wt[N][K] bf16
    int idx = blockIdx.x * 256 + threadIdx.x;
    if (idx >= K * N) return;
    int k = idx / N, n = idx - k * N;
    Wt[(size_t)n * K + k] = f2bf(W[(size_t)k * N + n]);
}

__global__ void fill_kernel(float* p, float v, int n) {
    int i = blockIdx.x * 256 + threadIdx.x;
    if (i < n) p[i] = v;
}

__global__ void embed_concat_kernel(const int* __restrict__ ui, const int* __restrict__ ii,
                                    const float* __restrict__ eu, const float* __restrict__ ei,
                                    float* __restrict__ x0, int n) {
    int idx = blockIdx.x * 256 + threadIdx.x;
    if (idx >= n * 128) return;
    int i = idx >> 7, d = idx & 127;
    float v = (d < 64) ? eu[(size_t)ui[i] * 64 + d] : ei[(size_t)ii[i] * 64 + (d - 64)];
    x0[(size_t)i * 128 + d] = v;
}

__global__ void degree_kernel(const int* __restrict__ dst, int ne, float* cnt) {
    int e = blockIdx.x * 256 + threadIdx.x;
    if (e < ne) atomicAdd(&cnt[dst[e]], 1.0f);
}

__global__ void dinv_kernel(const float* __restrict__ cnt, float* dinv, float* dinvc,
                            float* selfc, int n) {
    int i = blockIdx.x * 256 + threadIdx.x;
    if (i >= n) return;
    float c = cnt[i];
    float dv = rsqrtf(c + 1.f);
    dinv[i] = dv;
    selfc[i] = dv * dv;
    dinvc[i] = (c > 0.f) ? rsqrtf(c) : 0.f;
}

// agg[i][d] = xw[i][d]*selfc[i] + bias[d]   (grid = n blocks, 256 threads)
__global__ void gcn_init_kernel(const float* __restrict__ xw, const float* __restrict__ selfc,
                                const float* __restrict__ bias, float* __restrict__ agg) {
    int i = blockIdx.x, d = threadIdx.x;
    agg[(size_t)i * 256 + d] = xw[(size_t)i * 256 + d] * selfc[i] + bias[d];
}

// mode 0: c = dinv[s]*dinv[t]; mode 1: c = 1; mode 2: c = -dinv[s]*dinv[t]
__global__ void edge_scatter_kernel(const int* __restrict__ src, const int* __restrict__ dst,
                                    const float* __restrict__ xw, const float* __restrict__ dinv,
                                    float* __restrict__ agg, int mode) {
    int e = blockIdx.x, d = threadIdx.x;
    int s = src[e], t = dst[e];
    float c = 1.f;
    if (mode == 0) c =  dinv[s] * dinv[t];
    if (mode == 2) c = -dinv[s] * dinv[t];
    atomicAdd(&agg[(size_t)t * 256 + d], xw[(size_t)s * 256 + d] * c);
}

__global__ void bn_relu_kernel(const float* __restrict__ agg, const float* __restrict__ g,
                               const float* __restrict__ bt, float* __restrict__ out, int n) {
    int d = blockIdx.x, t = threadIdx.x;
    __shared__ float ssum[256], ssq[256];
    float s = 0.f, q = 0.f;
    for (int i = t; i < n; i += 256) {
        float v = agg[(size_t)i * 256 + d];
        s += v; q += v * v;
    }
    ssum[t] = s; ssq[t] = q; __syncthreads();
    for (int k = 128; k > 0; k >>= 1) {
        if (t < k) { ssum[t] += ssum[t + k]; ssq[t] += ssq[t + k]; }
        __syncthreads();
    }
    float mu  = ssum[0] / n;
    float var = ssq[0] / n - mu * mu;
    float scale = g[d] * rsqrtf(var + 1e-5f);
    float shift = bt[d] - mu * scale;
    for (int i = t; i < n; i += 256) {
        float v = agg[(size_t)i * 256 + d];
        out[(size_t)i * 256 + d] = fmaxf(v * scale + shift, 0.f);
    }
}

__global__ void sage_div_kernel(float* __restrict__ agg, const float* __restrict__ cnt) {
    int i = blockIdx.x, d = threadIdx.x;
    agg[(size_t)i * 256 + d] /= fmaxf(cnt[i], 1.f);
}

__global__ void gat_dots_kernel(const float* __restrict__ xw, const float* __restrict__ as,
                                const float* __restrict__ ad, float* asx, float* adx) {
    int i = blockIdx.x, t = threadIdx.x;
    __shared__ float s1[256], s2[256];
    float v = xw[(size_t)i * 256 + t];
    s1[t] = v * as[t]; s2[t] = v * ad[t];
    __syncthreads();
    for (int k = 128; k > 0; k >>= 1) {
        if (t < k) { s1[t] += s1[t + k]; s2[t] += s2[t + k]; }
        __syncthreads();
    }
    if (t == 0) { asx[i] = s1[0]; adx[i] = s2[0]; }
}

__global__ void gat_max_kernel(const int* __restrict__ src, const int* __restrict__ dst, int E_,
                               int n, const float* __restrict__ asx, const float* __restrict__ adx,
                               float* m) {
    int e = blockIdx.x * 256 + threadIdx.x;
    if (e >= E_ + n) return;
    int s = (e < E_) ? src[e] : (e - E_);
    int d = (e < E_) ? dst[e] : (e - E_);
    atomicMaxFloat(&m[d], leaky02(asx[s] + adx[d]));
}

__global__ void gat_denom_kernel(const int* __restrict__ src, const int* __restrict__ dst, int E_,
                                 int n, const float* __restrict__ asx, const float* __restrict__ adx,
                                 const float* __restrict__ m, float* den) {
    int e = blockIdx.x * 256 + threadIdx.x;
    if (e >= E_ + n) return;
    int s = (e < E_) ? src[e] : (e - E_);
    int d = (e < E_) ? dst[e] : (e - E_);
    atomicAdd(&den[d], expf(leaky02(asx[s] + adx[d]) - m[d]));
}

__global__ void gat_scatter_kernel(const int* __restrict__ src, const int* __restrict__ dst, int E_,
                                   const float* __restrict__ asx, const float* __restrict__ adx,
                                   const float* __restrict__ m, const float* __restrict__ den,
                                   const float* __restrict__ xw, float* __restrict__ agg) {
    int e = blockIdx.x, d = threadIdx.x;
    int s = (e < E_) ? src[e] : (e - E_);
    int t = (e < E_) ? dst[e] : (e - E_);
    float alpha = expf(leaky02(asx[s] + adx[t]) - m[t]) / den[t];
    atomicAdd(&agg[(size_t)t * 256 + d], xw[(size_t)s * 256 + d] * alpha);
}

__global__ void elu_bias_kernel(const float* __restrict__ agg, const float* __restrict__ b,
                                float* __restrict__ out) {
    int i = blockIdx.x, d = threadIdx.x;
    float v = agg[(size_t)i * 256 + d] + b[d];
    out[(size_t)i * 256 + d] = (v > 0.f) ? v : (expf(v) - 1.f);
}

// ---------------- host ----------------
extern "C" void kernel_launch(void* const* d_in, const int* in_sizes, int n_in,
                              void* d_out, int out_size, void* d_ws, size_t ws_size,
                              hipStream_t stream) {
    const int* ui   = (const int*)d_in[0];
    const int* ii   = (const int*)d_in[1];
    const int* eidx = (const int*)d_in[2];
    const float* emb_user = (const float*)d_in[3];
    const float* emb_item = (const float*)d_in[4];
    const float* mlp_w1 = (const float*)d_in[5];  const float* mlp_b1 = (const float*)d_in[6];
    const float* mlp_w2 = (const float*)d_in[7];  const float* mlp_b2 = (const float*)d_in[8];
    const float* gcn_w0 = (const float*)d_in[9];  const float* gcn_b0 = (const float*)d_in[10];
    const float* bn_g0  = (const float*)d_in[11]; const float* bn_b0  = (const float*)d_in[12];
    const float* gcn_w1 = (const float*)d_in[13]; const float* gcn_b1 = (const float*)d_in[14];
    const float* bn_g1  = (const float*)d_in[15]; const float* bn_b1  = (const float*)d_in[16];
    const float* sage_wl = (const float*)d_in[17]; const float* sage_bl = (const float*)d_in[18];
    const float* sage_wr = (const float*)d_in[19];
    const float* cheb_w0 = (const float*)d_in[20]; const float* cheb_w1 = (const float*)d_in[21];
    const float* cheb_b  = (const float*)d_in[22];
    const float* gat_w0 = (const float*)d_in[23]; const float* gat_as0 = (const float*)d_in[24];
    const float* gat_ad0 = (const float*)d_in[25]; const float* gat_b0 = (const float*)d_in[26];
    const float* gat_w1 = (const float*)d_in[27]; const float* gat_as1 = (const float*)d_in[28];
    const float* gat_ad1 = (const float*)d_in[29]; const float* gat_b1 = (const float*)d_in[30];
    const float* pred_w = (const float*)d_in[31]; const float* pred_b = (const float*)d_in[32];

    const int Nn   = in_sizes[0];          // 4096
    const int Ee   = in_sizes[2] / 2;      // 131072
    const int HID1 = in_sizes[6];          // 1024
    const int HID2 = in_sizes[8];          // 512
    const int D    = in_sizes[10];         // 256
    const int NC   = in_sizes[32];         // 41476
    const int* esrc = eidx;
    const int* edst = eidx + Ee;

    char* wsb = (char*)d_ws;
    size_t off = 0;
    auto alloc = [&](size_t bytes) -> void* {
        void* p = wsb + off;
        off += (bytes + 255) & ~(size_t)255;
        return p;
    };
    unsigned short* wt_mlp1  = (unsigned short*)alloc((size_t)128 * HID1 * 2);
    unsigned short* wt_mlp2  = (unsigned short*)alloc((size_t)HID1 * HID2 * 2);
    unsigned short* wt_gcn0  = (unsigned short*)alloc((size_t)HID2 * D * 2);
    unsigned short* wt_gcn1  = (unsigned short*)alloc((size_t)D * D * 2);
    unsigned short* wt_sagel = (unsigned short*)alloc((size_t)D * D * 2);
    unsigned short* wt_sager = (unsigned short*)alloc((size_t)D * D * 2);
    unsigned short* wt_cheb0 = (unsigned short*)alloc((size_t)D * D * 2);
    unsigned short* wt_cheb1 = (unsigned short*)alloc((size_t)D * D * 2);
    unsigned short* wt_gat0  = (unsigned short*)alloc((size_t)D * D * 2);
    unsigned short* wt_gat1  = (unsigned short*)alloc((size_t)D * D * 2);
    unsigned short* wt_pred  = (unsigned short*)alloc((size_t)NC * D * 2);
    float* X0  = (float*)alloc((size_t)Nn * 128 * 4);
    float* H1  = (float*)alloc((size_t)Nn * HID1 * 4);
    float* H2  = (float*)alloc((size_t)Nn * HID2 * 4);
    float* XW  = (float*)alloc((size_t)Nn * D * 4);
    float* AGG = (float*)alloc((size_t)Nn * D * 4);
    float* XA  = (float*)alloc((size_t)Nn * D * 4);
    float* XB  = (float*)alloc((size_t)Nn * D * 4);
    float* cnt   = (float*)alloc((size_t)Nn * 4);
    float* dinv  = (float*)alloc((size_t)Nn * 4);
    float* dinvc = (float*)alloc((size_t)Nn * 4);
    float* selfc = (float*)alloc((size_t)Nn * 4);
    float* asx   = (float*)alloc((size_t)Nn * 4);
    float* adx   = (float*)alloc((size_t)Nn * 4);
    float* mmax  = (float*)alloc((size_t)Nn * 4);
    float* den   = (float*)alloc((size_t)Nn * 4);
    (void)ws_size; (void)n_in; (void)out_size;

    auto tcvt = [&](const float* W, unsigned short* Wt, int K, int N) {
        transpose_cvt_kernel<<<((size_t)K * N + 255) / 256, 256, 0, stream>>>(W, Wt, K, N);
    };
    auto gemm = [&](const float* A, const unsigned short* Bt, const float* bias,
                    const float* addend, float* C, int M, int N, int K, int act) {
        dim3 g((N + 127) / 128, M / 128);
        gemm_bf16_wmma<<<g, 256, 0, stream>>>(A, Bt, bias, addend, C, M, N, K, act);
    };
    auto fill = [&](float* p, float v, int n) {
        fill_kernel<<<(n + 255) / 256, 256, 0, stream>>>(p, v, n);
    };

    // 0) weight prep: fp32 [K,N] -> bf16 [N,K]
    tcvt(mlp_w1, wt_mlp1, 128, HID1);
    tcvt(mlp_w2, wt_mlp2, HID1, HID2);
    tcvt(gcn_w0, wt_gcn0, HID2, D);
    tcvt(gcn_w1, wt_gcn1, D, D);
    tcvt(sage_wl, wt_sagel, D, D);
    tcvt(sage_wr, wt_sager, D, D);
    tcvt(cheb_w0, wt_cheb0, D, D);
    tcvt(cheb_w1, wt_cheb1, D, D);
    tcvt(gat_w0, wt_gat0, D, D);
    tcvt(gat_w1, wt_gat1, D, D);
    tcvt(pred_w, wt_pred, D, NC);

    // 1) embeddings + MLP
    embed_concat_kernel<<<((size_t)Nn * 128 + 255) / 256, 256, 0, stream>>>(
        ui, ii, emb_user, emb_item, X0, Nn);
    gemm(X0, wt_mlp1, mlp_b1, nullptr, H1, Nn, HID1, 128, 1);
    gemm(H1, wt_mlp2, mlp_b2, nullptr, H2, Nn, HID2, HID1, 1);

    // 2) degrees
    fill(cnt, 0.f, Nn);
    degree_kernel<<<(Ee + 255) / 256, 256, 0, stream>>>(edst, Ee, cnt);
    dinv_kernel<<<(Nn + 255) / 256, 256, 0, stream>>>(cnt, dinv, dinvc, selfc, Nn);

    // 3) GCN layer 0: H2 -> XA
    gemm(H2, wt_gcn0, nullptr, nullptr, XW, Nn, D, HID2, 0);
    gcn_init_kernel<<<Nn, 256, 0, stream>>>(XW, selfc, gcn_b0, AGG);
    edge_scatter_kernel<<<Ee, 256, 0, stream>>>(esrc, edst, XW, dinv, AGG, 0);
    bn_relu_kernel<<<D, 256, 0, stream>>>(AGG, bn_g0, bn_b0, XA, Nn);

    // 4) GCN layer 1: XA -> XB
    gemm(XA, wt_gcn1, nullptr, nullptr, XW, Nn, D, D, 0);
    gcn_init_kernel<<<Nn, 256, 0, stream>>>(XW, selfc, gcn_b1, AGG);
    edge_scatter_kernel<<<Ee, 256, 0, stream>>>(esrc, edst, XW, dinv, AGG, 0);
    bn_relu_kernel<<<D, 256, 0, stream>>>(AGG, bn_g1, bn_b1, XB, Nn);

    // 5) SAGE: XB -> XA
    fill(AGG, 0.f, Nn * D);
    edge_scatter_kernel<<<Ee, 256, 0, stream>>>(esrc, edst, XB, dinv, AGG, 1);
    sage_div_kernel<<<Nn, 256, 0, stream>>>(AGG, cnt);
    gemm(XB, wt_sager, nullptr, nullptr, XW, Nn, D, D, 0);
    gemm(AGG, wt_sagel, sage_bl, XW, XA, Nn, D, D, 1);

    // 6) Cheb K=2: XA -> XB
    fill(AGG, 0.f, Nn * D);
    edge_scatter_kernel<<<Ee, 256, 0, stream>>>(esrc, edst, XA, dinvc, AGG, 2);
    gemm(AGG, wt_cheb1, nullptr, nullptr, XW, Nn, D, D, 0);
    gemm(XA, wt_cheb0, cheb_b, XW, XB, Nn, D, D, 1);

    // 7) GAT layers (with self loops)
    const int tot = Ee + Nn;
    // layer 0: XB -> XA
    gemm(XB, wt_gat0, nullptr, nullptr, XW, Nn, D, D, 0);
    gat_dots_kernel<<<Nn, 256, 0, stream>>>(XW, gat_as0, gat_ad0, asx, adx);
    fill(mmax, -1e30f, Nn);
    fill(den, 0.f, Nn);
    fill(AGG, 0.f, Nn * D);
    gat_max_kernel<<<(tot + 255) / 256, 256, 0, stream>>>(esrc, edst, Ee, Nn, asx, adx, mmax);
    gat_denom_kernel<<<(tot + 255) / 256, 256, 0, stream>>>(esrc, edst, Ee, Nn, asx, adx, mmax, den);
    gat_scatter_kernel<<<tot, 256, 0, stream>>>(esrc, edst, Ee, asx, adx, mmax, den, XW, AGG);
    elu_bias_kernel<<<Nn, 256, 0, stream>>>(AGG, gat_b0, XA);
    // layer 1: XA -> XB
    gemm(XA, wt_gat1, nullptr, nullptr, XW, Nn, D, D, 0);
    gat_dots_kernel<<<Nn, 256, 0, stream>>>(XW, gat_as1, gat_ad1, asx, adx);
    fill(mmax, -1e30f, Nn);
    fill(den, 0.f, Nn);
    fill(AGG, 0.f, Nn * D);
    gat_max_kernel<<<(tot + 255) / 256, 256, 0, stream>>>(esrc, edst, Ee, Nn, asx, adx, mmax);
    gat_denom_kernel<<<(tot + 255) / 256, 256, 0, stream>>>(esrc, edst, Ee, Nn, asx, adx, mmax, den);
    gat_scatter_kernel<<<tot, 256, 0, stream>>>(esrc, edst, Ee, asx, adx, mmax, den, XW, AGG);
    elu_bias_kernel<<<Nn, 256, 0, stream>>>(AGG, gat_b1, XB);

    // 8) prediction: XB[4096,256] @ pred_w[256,41476] + pred_b -> d_out
    gemm(XB, wt_pred, pred_b, nullptr, (float*)d_out, Nn, NC, D, 0);
}
